// GATEdgePredictor_85744727097863
// MI455X (gfx1250) — matrix operand order repaired and checked by reading, not compiled, not run on hardware
//
#include <hip/hip_runtime.h>
#include <hip/hip_bf16.h>

// ---------------------------------------------------------------------------
// CDNA5 (gfx1250) GAT edge predictor.
// GEMMs use v_wmma_f32_16x16x32_f16 (wave32); each wave computes a 16x64
// output strip (4 WMMAs per K-step sharing one A fragment) to cut A traffic.
// Scatter/gather softmax aggregation uses f32 global atomics (bandwidth bound).
// ---------------------------------------------------------------------------

typedef _Float16 v16h __attribute__((ext_vector_type(16)));
typedef float    v8f  __attribute__((ext_vector_type(8)));
typedef _Float16 vh2  __attribute__((ext_vector_type(2)));
typedef float    vf2  __attribute__((ext_vector_type(2)));

// A-matrix (16x32 f16) K-pair base for VGPR-half v (0..7), lane group g (0..1)
__device__ __forceinline__ int kbaseA(int v, int g) {
    return ((v & 3) * 2) + (g * 8) + ((v >> 2) * 16);
}
// B-matrix (32x16 f16) K-pair base, same K order as A (v<4: K=8g+2v, v>=4: K=16+8g+2(v-4))
__device__ __forceinline__ int kbaseB(int v, int g) {
    return ((v & 3) * 2) + (g * 8) + ((v >> 2) * 16);
}

// ---------------- transpose + f32->f16 convert of weight matrices -----------
// W is [K, O] row-major f32; Wt becomes [O, K] row-major f16.
__global__ void conv_wt_kernel(const float* __restrict__ W, _Float16* __restrict__ Wt,
                               int K, int O) {
    int i = blockIdx.x * blockDim.x + threadIdx.x;
    if (i >= O * K) return;
    int o = i / K, k = i % K;
    Wt[i] = (_Float16)W[(size_t)k * O + o];
}

// ---------------- generic fill ---------------------------------------------
__global__ void fill_f32_kernel(float* __restrict__ p, float v, int n) {
    int i = blockIdx.x * blockDim.x + threadIdx.x;
    if (i < n) p[i] = v;
}

// ---------------- WMMA GEMM: C[M,Ncols] = A[M,K](f32->f16) x Bt[Ncols,K](f16)
// one wave per 16x64 output strip (4 n-tiles sharing the A fragment);
// M % 16 == 0, K % 32 == 0, Ncols % 64 == 0
__global__ void gemm_f16_wmma_kernel(const float* __restrict__ A,
                                     const _Float16* __restrict__ Bt,
                                     float* __restrict__ C,
                                     int M, int K, int Ncols) {
    int wave = (blockIdx.x * blockDim.x + threadIdx.x) >> 5;
    int lane = threadIdx.x & 31;
    int ngrp = Ncols >> 6;                    // number of 64-wide column groups
    int mt = wave / ngrp;
    int ng = wave % ngrp;
    if (mt * 16 >= M) return;                 // wave-uniform guard
    int mrow = mt * 16 + (lane & 15);         // A row this lane feeds
    int g = lane >> 4;

    v8f acc[4] = {};
    for (int k0 = 0; k0 < K; k0 += 32) {
        const float* arow = A + (size_t)mrow * K + k0;
        v16h a;
#pragma unroll
        for (int v = 0; v < 8; ++v) {
            int ka = kbaseA(v, g);
            vf2 af = *(const vf2*)(arow + ka);     // 8B-aligned (ka even)
            a[2 * v]     = (_Float16)af.x;
            a[2 * v + 1] = (_Float16)af.y;
        }
#pragma unroll
        for (int t = 0; t < 4; ++t) {
            int ncol = ng * 64 + t * 16 + (lane & 15);   // B column this lane feeds
            const _Float16* brow = Bt + (size_t)ncol * K + k0;
            v16h b;
#pragma unroll
            for (int v = 0; v < 8; ++v) {
                int kb = kbaseB(v, g);
                vh2 bh = *(const vh2*)(brow + kb);       // 4B-aligned
                b[2 * v]     = bh.x;
                b[2 * v + 1] = bh.y;
            }
            acc[t] = __builtin_amdgcn_wmma_f32_16x16x32_f16(false, a, false, b,
                                                            (short)0, acc[t], false, false);
        }
    }
    // C/D layout: VGPR r -> row r + 8*g; col = lane&15 within each 16-wide tile
#pragma unroll
    for (int t = 0; t < 4; ++t) {
#pragma unroll
        for (int r = 0; r < 8; ++r) {
            int row = mt * 16 + r + 8 * g;
            C[(size_t)row * Ncols + ng * 64 + t * 16 + (lane & 15)] = acc[t][r];
        }
    }
}

// ---------------- attention terms: als/ald [N,H] ---------------------------
// one wave per (node, head); 64 channels per head
__global__ void attn_terms_kernel(const float* __restrict__ t,
                                  const float* __restrict__ a_s,
                                  const float* __restrict__ a_d,
                                  float* __restrict__ als, float* __restrict__ ald,
                                  int Nn, int H) {
    int w = (blockIdx.x * blockDim.x + threadIdx.x) >> 5;
    int lane = threadIdx.x & 31;
    if (w >= Nn * H) return;
    int n = w / H, h = w % H;
    int Ncols = H * 64;
    const float* row = t + (size_t)n * Ncols + h * 64;
    float ps = row[lane] * a_s[h * 64 + lane] + row[lane + 32] * a_s[h * 64 + lane + 32];
    float pd = row[lane] * a_d[h * 64 + lane] + row[lane + 32] * a_d[h * 64 + lane + 32];
#pragma unroll
    for (int o = 16; o > 0; o >>= 1) {
        ps += __shfl_xor(ps, o, 32);
        pd += __shfl_xor(pd, o, 32);
    }
    if (lane == 0) { als[w] = ps; ald[w] = pd; }
}

// ---------------- CAS-based float atomic max -------------------------------
__device__ __forceinline__ void atomicMaxF(float* addr, float val) {
    unsigned int* ua = (unsigned int*)addr;
    unsigned int old = *ua;
    while (__uint_as_float(old) < val) {
        unsigned int assumed = old;
        old = atomicCAS(ua, assumed, __float_as_uint(val));
        if (old == assumed) break;
    }
}

__device__ __forceinline__ float leaky02(float x) { return x > 0.f ? x : 0.2f * x; }

// ---------------- segment max of edge logits -------------------------------
__global__ void edge_max_kernel(const int* __restrict__ src, const int* __restrict__ dst,
                                int E0, int Etot,
                                const float* __restrict__ als, const float* __restrict__ ald,
                                float* __restrict__ m, int H) {
    int i = blockIdx.x * blockDim.x + threadIdx.x;
    if (i >= Etot * H) return;
    int e = i / H, h = i % H;
    int s = (e < E0) ? src[e] : (e - E0);
    int d = (e < E0) ? dst[e] : (e - E0);
    float x = leaky02(als[s * H + h] + ald[d * H + h]);
    atomicMaxF(&m[d * H + h], x);
}

// ---------------- segment sum of exp(e - m) --------------------------------
__global__ void edge_denom_kernel(const int* __restrict__ src, const int* __restrict__ dst,
                                  int E0, int Etot,
                                  const float* __restrict__ als, const float* __restrict__ ald,
                                  const float* __restrict__ m, float* __restrict__ den, int H) {
    int i = blockIdx.x * blockDim.x + threadIdx.x;
    if (i >= Etot * H) return;
    int e = i / H, h = i % H;
    int s = (e < E0) ? src[e] : (e - E0);
    int d = (e < E0) ? dst[e] : (e - E0);
    float x = leaky02(als[s * H + h] + ald[d * H + h]);
    atomicAdd(&den[d * H + h], __expf(x - m[d * H + h]));
}

// ---------------- weighted scatter-aggregate -------------------------------
// one wave per (edge, head); lane l handles channels l and l+32
__global__ void edge_aggregate_kernel(const int* __restrict__ src, const int* __restrict__ dst,
                                      int E0, int Etot,
                                      const float* __restrict__ als, const float* __restrict__ ald,
                                      const float* __restrict__ m, const float* __restrict__ den,
                                      const float* __restrict__ t, float* __restrict__ agg, int H) {
    int w = (blockIdx.x * blockDim.x + threadIdx.x) >> 5;
    int lane = threadIdx.x & 31;
    if (w >= Etot * H) return;
    int e = w / H, h = w % H;
    int s = (e < E0) ? src[e] : (e - E0);
    int d = (e < E0) ? dst[e] : (e - E0);
    float x = leaky02(als[s * H + h] + ald[d * H + h]);
    float alpha = __expf(x - m[d * H + h]) / (den[d * H + h] + 1e-16f);
    int Ncols = H * 64;
    const float* trow = t + (size_t)s * Ncols + h * 64;
    float* arow = agg + (size_t)d * Ncols + h * 64;
    atomicAdd(&arow[lane],      alpha * trow[lane]);
    atomicAdd(&arow[lane + 32], alpha * trow[lane + 32]);
}

// ---------------- out = relu(agg + bias) (in place OK) ---------------------
__global__ void finalize_kernel(const float* __restrict__ agg, const float* __restrict__ b,
                                float* __restrict__ hout, int total, int C) {
    int i = blockIdx.x * blockDim.x + threadIdx.x;
    if (i >= total) return;
    float v = agg[i] + b[i % C];
    hout[i] = v > 0.f ? v : 0.f;
}

// ---------------- fused edge MLP -------------------------------------------
// A row = [h3[src] | h3[dst]] (K=128), W1t f16 [64][128], then relu, dot mW2,
// sigmoid. One wave handles 16 edges with 4 column-tile WMMA accumulators.
__global__ void edge_mlp_kernel(const int* __restrict__ src, const int* __restrict__ dst,
                                const float* __restrict__ h3,
                                const _Float16* __restrict__ w1t,
                                const float* __restrict__ b1,
                                const float* __restrict__ w2,
                                const float* __restrict__ b2,
                                float* __restrict__ out) {
    int wave = (blockIdx.x * blockDim.x + threadIdx.x) >> 5;  // edge tile id
    int lane = threadIdx.x & 31;
    int mlane = lane & 15;
    int g = lane >> 4;
    int e = wave * 16 + mlane;                 // edge whose A-row this lane feeds
    int s = src[e], d = dst[e];
    const float* srow = h3 + (size_t)s * 64;
    const float* drow = h3 + (size_t)d * 64;

    v8f acc[4] = {};
    for (int k0 = 0; k0 < 128; k0 += 32) {
        v16h a;
#pragma unroll
        for (int v = 0; v < 8; ++v) {
            int k = kbaseA(v, g) + k0;
            const float* p = (k < 64) ? (srow + k) : (drow + (k - 64));
            vf2 af = *(const vf2*)p;
            a[2 * v]     = (_Float16)af.x;
            a[2 * v + 1] = (_Float16)af.y;
        }
#pragma unroll
        for (int t = 0; t < 4; ++t) {
            const _Float16* brow = w1t + (size_t)(t * 16 + mlane) * 128 + k0;
            v16h b;
#pragma unroll
            for (int v = 0; v < 8; ++v) {
                int kb = kbaseB(v, g);
                vh2 bh = *(const vh2*)(brow + kb);
                b[2 * v]     = bh.x;
                b[2 * v + 1] = bh.y;
            }
            acc[t] = __builtin_amdgcn_wmma_f32_16x16x32_f16(false, a, false, b,
                                                            (short)0, acc[t], false, false);
        }
    }
    // second layer: z = relu(acc + b1), partial = sum_cols z * w2 for each row r
    float rs[8];
#pragma unroll
    for (int r = 0; r < 8; ++r) {
        float p = 0.f;
#pragma unroll
        for (int t = 0; t < 4; ++t) {
            int col = t * 16 + mlane;
            float z = acc[t][r] + b1[col];
            z = z > 0.f ? z : 0.f;
            p += z * w2[col];
        }
        rs[r] = p;
    }
    // reduce across the 16 lanes of each half-wave group
#pragma unroll
    for (int o = 1; o < 16; o <<= 1) {
#pragma unroll
        for (int r = 0; r < 8; ++r) rs[r] += __shfl_xor(rs[r], o, 32);
    }
    // group g holds rows 8g..8g+7 (row = r + 8g); lanes m<8 write row m+8g
    if (mlane < 8) {
        float z = rs[mlane] + b2[0];
        out[wave * 16 + 8 * g + mlane] = 1.f / (1.f + __expf(-z));
    }
}

// ---------------------------------------------------------------------------
static void run_gat_layer(const float* Ain, int K, const _Float16* Wt, int H,
                          const float* a_s, const float* a_d, const float* bias,
                          const int* src, const int* dst, int E0, int Etot, int Nn,
                          float* t_buf, float* h_out, float* als, float* ald,
                          float* mbuf, float* den, hipStream_t stream) {
    int Ncols = H * 64;
    int waves = (Nn / 16) * (Ncols / 64);      // one wave per 16x64 strip
    gemm_f16_wmma_kernel<<<(waves + 7) / 8, 256, 0, stream>>>(Ain, Wt, t_buf, Nn, K, Ncols);
    int wa = Nn * H;
    attn_terms_kernel<<<(wa + 7) / 8, 256, 0, stream>>>(t_buf, a_s, a_d, als, ald, Nn, H);
    fill_f32_kernel<<<(wa + 255) / 256, 256, 0, stream>>>(mbuf, -3e38f, wa);
    fill_f32_kernel<<<(wa + 255) / 256, 256, 0, stream>>>(den, 0.f, wa);
    int tot = Nn * Ncols;
    fill_f32_kernel<<<(tot + 255) / 256, 256, 0, stream>>>(h_out, 0.f, tot);
    int eh = Etot * H;
    edge_max_kernel<<<(eh + 255) / 256, 256, 0, stream>>>(src, dst, E0, Etot, als, ald, mbuf, H);
    edge_denom_kernel<<<(eh + 255) / 256, 256, 0, stream>>>(src, dst, E0, Etot, als, ald, mbuf, den, H);
    edge_aggregate_kernel<<<(eh + 7) / 8, 256, 0, stream>>>(src, dst, E0, Etot, als, ald, mbuf, den,
                                                            t_buf, h_out, H);
    finalize_kernel<<<(tot + 255) / 256, 256, 0, stream>>>(h_out, bias, h_out, tot, Ncols);
}

extern "C" void kernel_launch(void* const* d_in, const int* in_sizes, int n_in,
                              void* d_out, int out_size, void* d_ws, size_t ws_size,
                              hipStream_t stream) {
    (void)in_sizes; (void)n_in; (void)out_size; (void)ws_size;
    const int N = 50000, E = 800000;
    const int Etot = E + N;

    const float* x   = (const float*)d_in[0];
    const int*   ei  = (const int*)d_in[1];
    const int*   src = ei;
    const int*   dst = ei + E;
    const float* W1  = (const float*)d_in[2];
    const float* as1 = (const float*)d_in[3];
    const float* ad1 = (const float*)d_in[4];
    const float* b1  = (const float*)d_in[5];
    const float* W2  = (const float*)d_in[6];
    const float* as2 = (const float*)d_in[7];
    const float* ad2 = (const float*)d_in[8];
    const float* b2  = (const float*)d_in[9];
    const float* W3  = (const float*)d_in[10];
    const float* as3 = (const float*)d_in[11];
    const float* ad3 = (const float*)d_in[12];
    const float* b3  = (const float*)d_in[13];
    const float* mW1 = (const float*)d_in[14];
    const float* mb1 = (const float*)d_in[15];
    const float* mW2 = (const float*)d_in[16];
    const float* mb2 = (const float*)d_in[17];
    float* out = (float*)d_out;

    // ---- workspace carve-up (offsets 256B-aligned) ----
    char* ws = (char*)d_ws;
    size_t off = 0;
    auto carve = [&](size_t bytes) {
        void* p = ws + off;
        off += (bytes + 255) & ~(size_t)255;
        return p;
    };
    float*    t_buf = (float*)carve((size_t)N * 256 * 4);   // GEMM output (pre-attention)
    float*    h_buf = (float*)carve((size_t)N * 256 * 4);   // aggregation / layer output
    float*    als   = (float*)carve((size_t)N * 4 * 4);
    float*    ald   = (float*)carve((size_t)N * 4 * 4);
    float*    mbuf  = (float*)carve((size_t)N * 4 * 4);
    float*    den   = (float*)carve((size_t)N * 4 * 4);
    _Float16* wt1   = (_Float16*)carve((size_t)64 * 128 * 2);
    _Float16* wt2   = (_Float16*)carve((size_t)256 * 64 * 2);
    _Float16* wt3   = (_Float16*)carve((size_t)64 * 256 * 2);
    _Float16* w1t   = (_Float16*)carve((size_t)64 * 128 * 2);

    // ---- weight transpose + convert to f16 ----
    conv_wt_kernel<<<(64 * 128 + 255) / 256, 256, 0, stream>>>(W1, wt1, 128, 64);
    conv_wt_kernel<<<(256 * 64 + 255) / 256, 256, 0, stream>>>(W2, wt2, 64, 256);
    conv_wt_kernel<<<(64 * 256 + 255) / 256, 256, 0, stream>>>(W3, wt3, 256, 64);
    conv_wt_kernel<<<(64 * 128 + 255) / 256, 256, 0, stream>>>(mW1, w1t, 128, 64);

    // ---- three GAT layers (h_buf holds layer output, also next layer input) ----
    run_gat_layer(x,     128, wt1, 1, as1, ad1, b1, src, dst, E, Etot, N,
                  t_buf, h_buf, als, ald, mbuf, den, stream);
    run_gat_layer(h_buf,  64, wt2, 4, as2, ad2, b2, src, dst, E, Etot, N,
                  t_buf, h_buf, als, ald, mbuf, den, stream);
    run_gat_layer(h_buf, 256, wt3, 1, as3, ad3, b3, src, dst, E, Etot, N,
                  t_buf, h_buf, als, ald, mbuf, den, stream);

    // ---- fused edge MLP: E/16 = 50000 waves ----
    edge_mlp_kernel<<<(E / 16) / 8, 256, 0, stream>>>(src, dst, h_buf, w1t,
                                                      mb1, mW2, mb2, out);
}
